// PointNetFeaturePropagationRes_33406255629153
// MI455X (gfx1250) — matrix operand order repaired and checked by reading, not compiled, not run on hardware
//
#include <hip/hip_runtime.h>
#include <hip/hip_bf16.h>
#include <float.h>

// ---------------------------------------------------------------------------
// PointNet Feature Propagation + residual MLP, CDNA5 (gfx1250, wave32).
// GEMMs via v_wmma_f32_16x16x32_bf16 (fp32 accumulate), BN folded to
// scale/shift, all three convs + residual fused in one kernel.
// ---------------------------------------------------------------------------

typedef __attribute__((ext_vector_type(16))) __bf16 v16bf;
typedef __attribute__((ext_vector_type(8)))  __bf16 v8bf;
typedef __attribute__((ext_vector_type(8)))  float  v8f;

#define WMMA_BF16(A, B, C) \
  __builtin_amdgcn_wmma_f32_16x16x32_bf16(false, (A), false, (B), (short)0, (C), false, false)

static constexpr int Bc   = 16;
static constexpr int Np   = 4096;
static constexpr int Sp   = 1024;
static constexpr int C1   = 128;
static constexpr int C2   = 256;
static constexpr int CIN  = 384;   // C1 + C2
static constexpr int COUT = 128;
static constexpr int Mrows = Bc * Np;   // 65536

// packed weight tile counts (16x16x32 fragments)
static constexpr int FUSE_TILES = (CIN / 32) * (COUT / 16);   // 12*8 = 96
static constexpr int C1_TILES   = (COUT / 32) * (COUT / 16);  // 4*8  = 32
static constexpr int C2_TILES   = C1_TILES;
// each tile = 32 lanes * 16 bf16 = 512 elements

// ---------------------------------------------------------------------------
// Fold BN into per-channel scale/shift:  y = s*(Wx) + t
//   s = g * rsqrt(v + 1e-5),  t = beta + s*(b_conv - m)
// scales layout (floats): [0:128)=s_fuse [128:256)=t_fuse
//                         [256:384)=s_bn1 [384:512)=t_bn1
//                         [512:640)=s_bn2 [640:768)=t_bn2
// ---------------------------------------------------------------------------
__global__ void fp_prep_scales(
    const float* fuse_b, const float* fg, const float* fbeta, const float* fm, const float* fv,
    const float* c1_b,   const float* g1, const float* b1,    const float* m1, const float* v1,
    const float* c2_b,   const float* g2, const float* b2,    const float* m2, const float* v2,
    float* sc)
{
  int o = threadIdx.x;
  if (o >= COUT) return;
  float s;
  s = fg[o] * rsqrtf(fv[o] + 1e-5f); sc[o]       = s; sc[128 + o] = fbeta[o] + s * (fuse_b[o] - fm[o]);
  s = g1[o] * rsqrtf(v1[o] + 1e-5f); sc[256 + o] = s; sc[384 + o] = b1[o]    + s * (c1_b[o]  - m1[o]);
  s = g2[o] * rsqrtf(v2[o] + 1e-5f); sc[512 + o] = s; sc[640 + o] = b2[o]    + s * (c2_b[o]  - m2[o]);
}

// ---------------------------------------------------------------------------
// Pack W[o][k] (f32) into per-lane B-fragment layout (bf16):
//   tile (kt,nt): lane l holds N = nt*16 + (l&15), K = kt*32 + ((l>>4)<<3) + e (e<8),
//   K = kt*32 + 16 + ((l>>4)<<3) + (e-8) (e>=8)
// 160 tiles * 32 lanes = 5120 threads.
// ---------------------------------------------------------------------------
__global__ void fp_pack_weights(const float* fuse_w, const float* c1_w, const float* c2_w,
                                __bf16* fuseP, __bf16* c1P, __bf16* c2P)
{
  int t = blockIdx.x * blockDim.x + threadIdx.x;
  if (t >= (FUSE_TILES + C1_TILES + C2_TILES) * 32) return;
  int tile = t >> 5, lane = t & 31;
  const float* src; __bf16* dst; int K, tl;
  if (tile < FUSE_TILES)              { tl = tile;                    src = fuse_w; dst = fuseP; K = CIN;  }
  else if (tile < FUSE_TILES + C1_TILES) { tl = tile - FUSE_TILES;       src = c1_w;  dst = c1P;  K = COUT; }
  else                                { tl = tile - FUSE_TILES - C1_TILES; src = c2_w; dst = c2P; K = COUT; }
  int kt = tl >> 3, nt = tl & 7;
  int o  = nt * 16 + (lane & 15);
  int kb = kt * 32 + ((lane >> 4) << 3);
  __bf16* d       = dst + ((size_t)tl * 32 + lane) * 16;
  const float* s  = src + (size_t)o * K + kb;
#pragma unroll
  for (int e = 0; e < 16; ++e) d[e] = (__bf16)s[e + ((e >> 3) << 3)];
}

// ---------------------------------------------------------------------------
// 3-NN inverse-distance interpolation; builds X (bf16, [Mrows][CIN] row-major):
//   X[:,0:128)   = points1^T
//   X[:,128:384) = interp(points2)
// One thread per point; xyz2 for the batch staged in LDS.
// ---------------------------------------------------------------------------
__global__ void fp_interp(const float* __restrict__ xyz1, const float* __restrict__ xyz2,
                          const float* __restrict__ points1, const float* __restrict__ points2,
                          __bf16* __restrict__ X)
{
  __shared__ float qx[Sp], qy[Sp], qz[Sp], qn[Sp];
  const int pts_per_blk = 128;
  int b  = blockIdx.x / (Np / pts_per_blk);
  int n0 = (blockIdx.x % (Np / pts_per_blk)) * pts_per_blk;

  for (int s = threadIdx.x; s < Sp; s += blockDim.x) {
    float x = xyz2[((size_t)b * Sp + s) * 3 + 0];
    float y = xyz2[((size_t)b * Sp + s) * 3 + 1];
    float z = xyz2[((size_t)b * Sp + s) * 3 + 2];
    qx[s] = x; qy[s] = y; qz[s] = z; qn[s] = x * x + y * y + z * z;
  }
  __syncthreads();

  int n = n0 + threadIdx.x;
  float px = xyz1[((size_t)b * Np + n) * 3 + 0];
  float py = xyz1[((size_t)b * Np + n) * 3 + 1];
  float pz = xyz1[((size_t)b * Np + n) * 3 + 2];
  float pn = px * px + py * py + pz * pz;

  float d0 = FLT_MAX, d1 = FLT_MAX, d2 = FLT_MAX;
  int   i0 = 0, i1 = 0, i2 = 0;
  for (int s = 0; s < Sp; ++s) {
    float d = pn + qn[s] - 2.0f * (px * qx[s] + py * qy[s] + pz * qz[s]);
    if (d < d2) {
      if (d < d1) {
        if (d < d0) { d2 = d1; i2 = i1; d1 = d0; i1 = i0; d0 = d; i0 = s; }
        else        { d2 = d1; i2 = i1; d1 = d;  i1 = s; }
      } else        { d2 = d;  i2 = s; }
    }
  }
  float r0 = 1.0f / (d0 + 1e-8f), r1 = 1.0f / (d1 + 1e-8f), r2 = 1.0f / (d2 + 1e-8f);
  float inv = 1.0f / (r0 + r1 + r2);
  float w0 = r0 * inv, w1 = r1 * inv, w2 = r2 * inv;

  __bf16* xr = X + ((size_t)b * Np + n) * CIN;
  union { v8bf v; __bf16 e[8]; } pk;

  const float* p1 = points1 + (size_t)b * C1 * Np;
#pragma unroll 2
  for (int c0 = 0; c0 < C1; c0 += 8) {
#pragma unroll
    for (int j = 0; j < 8; ++j) pk.e[j] = (__bf16)p1[(size_t)(c0 + j) * Np + n];
    *(v8bf*)(xr + c0) = pk.v;
  }
  const float* p2 = points2 + (size_t)b * C2 * Sp;
#pragma unroll 2
  for (int c0 = 0; c0 < C2; c0 += 8) {
#pragma unroll
    for (int j = 0; j < 8; ++j) {
      const float* pc = p2 + (size_t)(c0 + j) * Sp;
      pk.e[j] = (__bf16)(w0 * pc[i0] + w1 * pc[i1] + w2 * pc[i2]);
    }
    *(v8bf*)(xr + C1 + c0) = pk.v;
  }
}

// ---------------------------------------------------------------------------
// Fused GEMM chain. 128 threads = 4 waves; each wave owns 16 rows, computes
// all 128 output channels (8 C fragments), chains fuse -> c1 -> c2 through
// LDS (C-layout -> A-layout reshape), adds residual, ReLU, and stores the
// transposed [B,128,N] output with 32B-contiguous per-lane stores.
// ---------------------------------------------------------------------------
__global__ void __launch_bounds__(128)
fp_fused_mlp(const __bf16* __restrict__ X,
             const __bf16* __restrict__ fuseP,
             const __bf16* __restrict__ c1P,
             const __bf16* __restrict__ c2P,
             const float*  __restrict__ sc,
             float* __restrict__ out)
{
  constexpr int WAVES = 4;
  constexpr int LPAD  = 136;                 // 128 + 8: rotates LDS banks per row
  __shared__ __bf16 xt[WAVES][16][LPAD];
  __shared__ __bf16 ht[WAVES][16][LPAD];

  const int lane = threadIdx.x & 31;
  const int w    = threadIdx.x >> 5;
  const int rowbase = blockIdx.x * (WAVES * 16) + w * 16;
  const int m    = lane & 15;                // A row / C column selector
  const int hk   = (lane >> 4) * 8;          // A K-offset & C row-offset for hi half

  union AFrag { v16bf v; v8bf h[2]; };
  const v8f vzero = {0.f, 0.f, 0.f, 0.f, 0.f, 0.f, 0.f, 0.f};
  v8f acc[8];

  // ---------------- GEMM 1: fuse (K = 384) ----------------
#pragma unroll
  for (int nt = 0; nt < 8; ++nt) acc[nt] = vzero;
  const __bf16* xrow = X + (size_t)(rowbase + m) * CIN;
  __builtin_prefetch(xrow, 0, 0);            // global_prefetch_b8
#pragma unroll 4
  for (int kt = 0; kt < CIN / 32; ++kt) {
    AFrag a;
    a.h[0] = *(const v8bf*)(xrow + kt * 32 + hk);
    a.h[1] = *(const v8bf*)(xrow + kt * 32 + hk + 16);
#pragma unroll
    for (int nt = 0; nt < 8; ++nt) {
      v16bf bf = *(const v16bf*)(fuseP + ((size_t)(kt * 8 + nt) * 32 + lane) * 16);
      acc[nt] = WMMA_BF16(a.v, bf, acc[nt]);
    }
  }
  // epilogue: BN-fold + ReLU -> xt (bf16)
#pragma unroll
  for (int nt = 0; nt < 8; ++nt) {
    int o = nt * 16 + m;
    float s = sc[o], t = sc[128 + o];
#pragma unroll
    for (int i = 0; i < 8; ++i) {
      float v = s * acc[nt][i] + t;
      xt[w][hk + i][o] = (__bf16)(v > 0.f ? v : 0.f);
    }
  }
  __syncthreads();

  // ---------------- GEMM 2: c1 (K = 128) ----------------
#pragma unroll
  for (int nt = 0; nt < 8; ++nt) acc[nt] = vzero;
#pragma unroll
  for (int kt = 0; kt < COUT / 32; ++kt) {
    AFrag a;
    a.h[0] = *(const v8bf*)(&xt[w][m][kt * 32 + hk]);
    a.h[1] = *(const v8bf*)(&xt[w][m][kt * 32 + hk + 16]);
#pragma unroll
    for (int nt = 0; nt < 8; ++nt) {
      v16bf bf = *(const v16bf*)(c1P + ((size_t)(kt * 8 + nt) * 32 + lane) * 16);
      acc[nt] = WMMA_BF16(a.v, bf, acc[nt]);
    }
  }
#pragma unroll
  for (int nt = 0; nt < 8; ++nt) {
    int o = nt * 16 + m;
    float s = sc[256 + o], t = sc[384 + o];
#pragma unroll
    for (int i = 0; i < 8; ++i) {
      float v = s * acc[nt][i] + t;
      ht[w][hk + i][o] = (__bf16)(v > 0.f ? v : 0.f);
    }
  }
  __syncthreads();

  // ---------------- GEMM 3: c2 (K = 128) + residual + ReLU ----------------
#pragma unroll
  for (int nt = 0; nt < 8; ++nt) acc[nt] = vzero;
#pragma unroll
  for (int kt = 0; kt < COUT / 32; ++kt) {
    AFrag a;
    a.h[0] = *(const v8bf*)(&ht[w][m][kt * 32 + hk]);
    a.h[1] = *(const v8bf*)(&ht[w][m][kt * 32 + hk + 16]);
#pragma unroll
    for (int nt = 0; nt < 8; ++nt) {
      v16bf bf = *(const v16bf*)(c2P + ((size_t)(kt * 8 + nt) * 32 + lane) * 16);
      acc[nt] = WMMA_BF16(a.v, bf, acc[nt]);
    }
  }
  // output: rows (rowbase+hk .. +hk+7) are 8 consecutive n sharing channel o
  const int grow = rowbase + hk;
  const int bidx = grow >> 12;               // / 4096
  const int nidx = grow & (Np - 1);
#pragma unroll
  for (int nt = 0; nt < 8; ++nt) {
    int o = nt * 16 + m;
    float s = sc[512 + o], t = sc[640 + o];
    float tmp[8];
#pragma unroll
    for (int i = 0; i < 8; ++i) {
      float y = s * acc[nt][i] + t + (float)xt[w][hk + i][o];
      tmp[i] = y > 0.f ? y : 0.f;
    }
    float* op = out + (((size_t)bidx * COUT + o) << 12) + nidx;
    *(float4*)(op)     = make_float4(tmp[0], tmp[1], tmp[2], tmp[3]);
    *(float4*)(op + 4) = make_float4(tmp[4], tmp[5], tmp[6], tmp[7]);
  }
}

// ---------------------------------------------------------------------------
extern "C" void kernel_launch(void* const* d_in, const int* in_sizes, int n_in,
                              void* d_out, int out_size, void* d_ws, size_t ws_size,
                              hipStream_t stream) {
  const float* xyz1    = (const float*)d_in[0];
  const float* xyz2    = (const float*)d_in[1];
  const float* points1 = (const float*)d_in[2];
  const float* points2 = (const float*)d_in[3];
  const float* fuse_w  = (const float*)d_in[4];
  const float* fuse_b  = (const float*)d_in[5];
  const float* fuse_g  = (const float*)d_in[6];
  const float* fuse_bt = (const float*)d_in[7];
  const float* fuse_m  = (const float*)d_in[8];
  const float* fuse_v  = (const float*)d_in[9];
  const float* c1_w    = (const float*)d_in[10];
  const float* c1_b    = (const float*)d_in[11];
  const float* bn1_g   = (const float*)d_in[12];
  const float* bn1_b   = (const float*)d_in[13];
  const float* bn1_m   = (const float*)d_in[14];
  const float* bn1_v   = (const float*)d_in[15];
  const float* c2_w    = (const float*)d_in[16];
  const float* c2_b    = (const float*)d_in[17];
  const float* bn2_g   = (const float*)d_in[18];
  const float* bn2_b   = (const float*)d_in[19];
  const float* bn2_m   = (const float*)d_in[20];
  const float* bn2_v   = (const float*)d_in[21];
  float* out = (float*)d_out;

  // workspace carve-up (bf16 elements)
  __bf16* X     = (__bf16*)d_ws;
  size_t  off   = (size_t)Mrows * CIN;           // 25,165,824 elems
  __bf16* fuseP = X + off; off += (size_t)FUSE_TILES * 512;
  __bf16* c1P   = X + off; off += (size_t)C1_TILES * 512;
  __bf16* c2P   = X + off; off += (size_t)C2_TILES * 512;
  float*  sc    = (float*)(X + off);             // 768 floats

  fp_prep_scales<<<1, 128, 0, stream>>>(
      fuse_b, fuse_g, fuse_bt, fuse_m, fuse_v,
      c1_b,   bn1_g,  bn1_b,   bn1_m,  bn1_v,
      c2_b,   bn2_g,  bn2_b,   bn2_m,  bn2_v, sc);

  fp_pack_weights<<<20, 256, 0, stream>>>(fuse_w, c1_w, c2_w, fuseP, c1P, c2P);

  fp_interp<<<Mrows / 128, 128, 0, stream>>>(xyz1, xyz2, points1, points2, X);

  fp_fused_mlp<<<Mrows / 64, 128, 0, stream>>>(X, fuseP, c1P, c2P, sc, out);
}
